// StackedLstm_55868934586871
// MI455X (gfx1250) — matrix operand order repaired and checked by reading, not compiled, not run on hardware
//
#include <hip/hip_runtime.h>

// ---------------------------------------------------------------------------
// Stacked highway-LSTM for MI455X (gfx1250), wave32 + WMMA bf16.
//   [one-time] x -> bf16; Wi -> (6H,K) bf16 transposed; Ws -> (5H,K) bf16
//   per layer: big WMMA GEMM pi = X @ Wi (LDS double-buffered)
//              ONE persistent kernel runs all 512 recurrence steps:
//                - Ws slice (160KB) resident in LDS for the whole sequence
//                - h kept bf16 (WMMA A) + f32 shadow (carry/final state)
//                - 2-way K-split across wave quads + LDS reduce
//                - device-wide release/acquire barrier between steps
//   Staging uses GLOBAL_LOAD_ASYNC_TO_LDS_B128 (ASYNCcnt) when available.
// ---------------------------------------------------------------------------

typedef __attribute__((ext_vector_type(16))) __bf16 v16bf;
typedef __attribute__((ext_vector_type(8)))  float  v8f;
typedef __attribute__((ext_vector_type(4)))  int    v4i_;

#define WMMA_BF16(a, b, c) \
  __builtin_amdgcn_wmma_f32_16x16x32_bf16(false, (a), false, (b), (short)0, (c), false, false)

// ---- async global->LDS staging (gfx1250 ASYNCcnt path), guarded ----
#if defined(__has_builtin)
#if __has_builtin(__builtin_amdgcn_global_load_async_to_lds_b128) && \
    __has_builtin(__builtin_amdgcn_s_wait_asynccnt)
#define USE_ASYNC_LDS 1
#endif
#endif
#ifndef USE_ASYNC_LDS
#define USE_ASYNC_LDS 0
#endif

#if USE_ASYNC_LDS
typedef __attribute__((address_space(1))) v4i_* g4ptr_t;
typedef __attribute__((address_space(3))) v4i_* l4ptr_t;
__device__ __forceinline__ void async_copy16(void* lds_dst, const void* gsrc) {
  __builtin_amdgcn_global_load_async_to_lds_b128(
      (g4ptr_t)gsrc, (l4ptr_t)lds_dst, 0, 0);
}
__device__ __forceinline__ void async_wait() {
  __builtin_amdgcn_s_wait_asynccnt(0);
}
#else
__device__ __forceinline__ void async_copy16(void* lds_dst, const void* gsrc) {
  *(uint4*)lds_dst = *(const uint4*)gsrc;
}
__device__ __forceinline__ void async_wait() {}
#endif

__device__ __forceinline__ float sigmoidf_(float x) {
  return 1.0f / (1.0f + __expf(-x));
}

constexpr int Bq = 64, Tq = 512, Dq = 1024, Hq = 1024;
constexpr int NBLK = 64;                       // persistent grid size

// padded LDS row length (kills 16-way bank conflict at stride 1024*2B)
constexpr int KP = 1024 + 8;
constexpr size_t LDS_WL  = (size_t)5 * 16 * KP * 2;    // 165,120 B
constexpr size_t LDS_HA  = (size_t)64 * KP * 2;        // 132,096 B
constexpr size_t LDS_RED = (size_t)4 * 5 * 32 * 8 * 4; // 20,480 B
constexpr size_t SMEM_PERS = LDS_WL + LDS_HA + LDS_RED; // 317,696 B < 320KB

// ---------------------------------------------------------------------------
// one-time converters
// ---------------------------------------------------------------------------
__global__ __launch_bounds__(256) void cvt_bf16_kernel(
    const float* __restrict__ in, __bf16* __restrict__ out, size_t n) {
  size_t i = ((size_t)blockIdx.x * blockDim.x + threadIdx.x) * 4;
  if (i + 3 < n) {
    const float4 v = *(const float4*)&in[i];
    union { __bf16 b[4]; uint2 u; } pk;
    pk.b[0] = (__bf16)v.x; pk.b[1] = (__bf16)v.y;
    pk.b[2] = (__bf16)v.z; pk.b[3] = (__bf16)v.w;
    *(uint2*)&out[i] = pk.u;
  }
}

// out[n*K + k] = (bf16) in[k*N + n]
__global__ __launch_bounds__(256) void transpose_cvt_kernel(
    const float* __restrict__ in, __bf16* __restrict__ out, int K, int N) {
  size_t i = (size_t)blockIdx.x * blockDim.x + threadIdx.x;
  if (i < (size_t)K * N) {
    int k = (int)(i / N), n = (int)(i % N);
    out[(size_t)n * K + k] = (__bf16)in[i];
  }
}

// ---------------------------------------------------------------------------
// Big GEMM: C(M,N) = A(M,K) @ B(K,N); A row-major bf16, Bt = B^T (N,K) bf16.
// ---------------------------------------------------------------------------
__global__ __launch_bounds__(256) void lstm_gemm_kernel(
    const __bf16* __restrict__ A, const __bf16* __restrict__ Bt,
    float* __restrict__ C, int M, int N, int K) {
  __shared__ __bf16 As[2][128][32];
  __shared__ __bf16 Bs[2][128][32];

  const int tid    = threadIdx.x;
  const int lane   = tid & 31;
  const int wave   = tid >> 5;
  const int waveM  = (wave & 3) * 32;
  const int waveN  = (wave >> 2) * 64;
  const int blockM = blockIdx.y * 128;
  const int blockN = blockIdx.x * 128;
  const int mloc   = lane & 15;
  const int hi     = lane >> 4;

  auto stage = [&](int buf, int kt) {
#pragma unroll
    for (int e = tid * 8; e < 128 * 32; e += 256 * 8) {
      const int m = e >> 5, k = e & 31;
      async_copy16(&As[buf][m][k], &A[(size_t)(blockM + m) * K + kt + k]);
      async_copy16(&Bs[buf][m][k], &Bt[(size_t)(blockN + m) * K + kt + k]);
    }
  };

  v8f acc[2][4];
#pragma unroll
  for (int i = 0; i < 2; i++)
#pragma unroll
    for (int j = 0; j < 4; j++) acc[i][j] = (v8f){0, 0, 0, 0, 0, 0, 0, 0};

  stage(0, 0);
  for (int kt = 0; kt < K; kt += 32) {
    const int buf = (kt >> 5) & 1;
    async_wait();          // staged buffer for this iteration has landed
    __syncthreads();
    if (kt + 32 < K) stage(buf ^ 1, kt + 32);
    if (kt + 64 < K) {
      __builtin_prefetch(&A[(size_t)(blockM + (tid & 127)) * K + kt + 64], 0, 1);
      __builtin_prefetch(&Bt[(size_t)(blockN + (tid & 127)) * K + kt + 64], 0, 1);
    }

    v16bf af[2], bf[4];
#pragma unroll
    for (int v = 0; v < 8; v++) {
      const int ka = ((v & 4) << 2) + (hi << 3) + ((v & 3) << 1);
      const int kb = ((v & 4) << 2) + (hi << 3) + ((v & 3) << 1);
#pragma unroll
      for (int mi = 0; mi < 2; mi++) {
        const int row = waveM + mi * 16 + mloc;
        af[mi][2 * v]     = As[buf][row][ka];
        af[mi][2 * v + 1] = As[buf][row][ka + 1];
      }
#pragma unroll
      for (int ni = 0; ni < 4; ni++) {
        const int col = waveN + ni * 16 + mloc;
        bf[ni][2 * v]     = Bs[buf][col][kb];
        bf[ni][2 * v + 1] = Bs[buf][col][kb + 1];
      }
    }
#pragma unroll
    for (int mi = 0; mi < 2; mi++)
#pragma unroll
      for (int ni = 0; ni < 4; ni++)
        acc[mi][ni] = WMMA_BF16(af[mi], bf[ni], acc[mi][ni]);
  }

#pragma unroll
  for (int mi = 0; mi < 2; mi++)
#pragma unroll
    for (int ni = 0; ni < 4; ni++) {
      const int col  = blockN + waveN + ni * 16 + mloc;
      const int rowb = blockM + waveM + mi * 16 + hi * 8;
#pragma unroll
      for (int r = 0; r < 8; r++)
        C[(size_t)(rowb + r) * N + col] = acc[mi][ni][r];
    }
}

// ---------------------------------------------------------------------------
// Persistent recurrence kernel: one launch per layer.
// grid = 64 blocks (jt = 16-col tile of H), block = 256 thr = 8 waves.
// waves 0-3: K in [0,512), handle gates; waves 4-7: K in [512,1024).
// ---------------------------------------------------------------------------
__global__ __launch_bounds__(256) void lstm_persistent_kernel(
    __bf16* __restrict__ hbf,          // [2][B*H] bf16 double buffer
    float* __restrict__ hfp,           // [2][B*H] f32 double buffer
    float* __restrict__ c,             // [B*H]
    const float* __restrict__ pi,      // (B,T,6H) f32
    const __bf16* __restrict__ Wst,    // (5H,K) bf16
    const float* __restrict__ bs,
    const int* __restrict__ lengths,
    float* __restrict__ y_f32, __bf16* __restrict__ y_bf16,
    int* __restrict__ barrier_cnt) {
  constexpr int Tn = Tq, Hn = Hq, H6 = 6 * Hq;
  const size_t BH = (size_t)Bq * Hn;

  extern __shared__ char smem[];
  __bf16 (*Wl)[16][KP] = (__bf16(*)[16][KP])smem;                 // [gate][n][k]
  __bf16 (*Hall)[KP]   = (__bf16(*)[KP])(smem + LDS_WL);          // [b][k]
  float* red           = (float*)(smem + LDS_WL + LDS_HA);        // [4][5][32][8]

  const int tid  = threadIdx.x;
  const int lane = tid & 31;
  const int wave = tid >> 5;
  const int mt   = wave & 3;            // row tile of B
  const int ks   = wave >> 2;           // K-split half
  const int jt   = blockIdx.x;          // column tile of H
  const int nloc = lane & 15;
  const int hi   = lane >> 4;
  const int col  = jt * 16 + nloc;

  // ---- load Ws slice once (160KB) ----
  for (int ch = tid; ch < 5 * 16 * 1024 / 8; ch += 256) {
    const int e = ch * 8;
    const int g = e >> 14, n = (e >> 10) & 15, k = e & 1023;
    async_copy16(&Wl[g][n][k],
                 &Wst[((size_t)g * Hn + jt * 16 + n) * Hn + k]);
  }

  float bsv[5];
#pragma unroll
  for (int g = 0; g < 5; g++) bsv[g] = bs[g * Hn + col];

  for (int t = 0; t < Tn; t++) {
    const __bf16* hprev_b = hbf + (size_t)(t & 1) * BH;
    __bf16*       hnext_b = hbf + (size_t)((t + 1) & 1) * BH;
    const float*  hprev_f = hfp + (size_t)(t & 1) * BH;
    float*        hnext_f = hfp + (size_t)((t + 1) & 1) * BH;

    // stage full h_prev (128KB bf16) into LDS
    for (int ch = tid; ch < 64 * 1024 / 8; ch += 256) {
      const int e = ch * 8;
      const int m = e >> 10, k = e & 1023;
      async_copy16(&Hall[m][k], &hprev_b[(size_t)m * Hn + k]);
    }
    async_wait();
    __syncthreads();

    v8f acc[5];
#pragma unroll
    for (int g = 0; g < 5; g++) acc[g] = (v8f){0, 0, 0, 0, 0, 0, 0, 0};

    const int k0 = ks * 512;
    for (int kt = k0; kt < k0 + 512; kt += 32) {
      v16bf af;
#pragma unroll
      for (int v = 0; v < 8; v++) {
        const int ka = kt + ((v & 4) << 2) + (hi << 3) + ((v & 3) << 1);
        const int row = mt * 16 + nloc;
        af[2 * v]     = Hall[row][ka];
        af[2 * v + 1] = Hall[row][ka + 1];
      }
#pragma unroll
      for (int g = 0; g < 5; g++) {
        v16bf bfr;
#pragma unroll
        for (int v = 0; v < 8; v++) {
          const int kb = kt + ((v & 4) << 2) + (hi << 3) + ((v & 3) << 1);
          bfr[2 * v]     = Wl[g][nloc][kb];
          bfr[2 * v + 1] = Wl[g][nloc][kb + 1];
        }
        acc[g] = WMMA_BF16(af, bfr, acc[g]);
      }
    }

    // 2-way K reduction through LDS
    if (ks == 1) {
#pragma unroll
      for (int g = 0; g < 5; g++)
#pragma unroll
        for (int r = 0; r < 8; r++)
          red[((mt * 5 + g) * 32 + lane) * 8 + r] = acc[g][r];
    }
    __syncthreads();

    if (ks == 0) {
#pragma unroll
      for (int g = 0; g < 5; g++)
#pragma unroll
        for (int r = 0; r < 8; r++)
          acc[g][r] += red[((mt * 5 + g) * 32 + lane) * 8 + r];

      // gates + state update; lane owns rows (mt*16 + hi*8 + r), col
#pragma unroll
      for (int r = 0; r < 8; r++) {
        const int b = mt * 16 + hi * 8 + r;
        const size_t pib = ((size_t)b * Tn + t) * H6;
        const float i_g = sigmoidf_(pi[pib + 0 * Hn + col] + acc[0][r] + bsv[0]);
        const float f_g = sigmoidf_(pi[pib + 1 * Hn + col] + acc[1][r] + bsv[1]);
        const float m_i = tanhf    (pi[pib + 2 * Hn + col] + acc[2][r] + bsv[2]);
        const float o_g = sigmoidf_(pi[pib + 3 * Hn + col] + acc[3][r] + bsv[3]);
        const float hw  = sigmoidf_(pi[pib + 4 * Hn + col] + acc[4][r] + bsv[4]);
        const float cp = c[b * Hn + col];
        const float cn = i_g * m_i + f_g * cp;
        float out = o_g * tanhf(cn);
        out = hw * out + (1.0f - hw) * pi[pib + 5 * Hn + col];
        const bool valid = t < lengths[b];
        const float hv = valid ? out : hprev_f[b * Hn + col];
        hnext_f[b * Hn + col] = hv;
        hnext_b[b * Hn + col] = (__bf16)hv;
        c[b * Hn + col]       = valid ? cn : cp;
        const float yv = valid ? out : 0.0f;
        if (y_f32)  y_f32[((size_t)b * Tn + t) * Hn + col] = yv;
        if (y_bf16) y_bf16[((size_t)b * Tn + t) * Hn + col] = (__bf16)yv;
      }
      // prefetch next step's pi slice (advisory)
      if (t + 1 < Tn) {
        const size_t pin = ((size_t)(mt * 16 + hi * 8) * Tn + (t + 1)) * H6;
#pragma unroll
        for (int g = 0; g < 6; g++) __builtin_prefetch(&pi[pin + g * Hn + col], 0, 1);
      }
    }

    // ---- device-wide step barrier (release/acquire) ----
    __threadfence();
    __syncthreads();
    if (tid == 0) {
      __hip_atomic_fetch_add(barrier_cnt, 1, __ATOMIC_RELEASE, __HIP_MEMORY_SCOPE_AGENT);
      const int target = NBLK * (t + 1);
      while (__hip_atomic_load(barrier_cnt, __ATOMIC_ACQUIRE, __HIP_MEMORY_SCOPE_AGENT) < target)
        __builtin_amdgcn_s_sleep(2);
    }
    __syncthreads();
    __threadfence();
  }
}

__global__ void zero_kernel(float* p, size_t n) {
  size_t i = (size_t)blockIdx.x * blockDim.x + threadIdx.x;
  if (i < n) p[i] = 0.0f;
}

__global__ void copy_kernel(const float* __restrict__ s, float* __restrict__ d, int n) {
  int i = blockIdx.x * blockDim.x + threadIdx.x;
  if (i < n) d[i] = s[i];
}

extern "C" void kernel_launch(void* const* d_in, const int* in_sizes, int n_in,
                              void* d_out, int out_size, void* d_ws, size_t ws_size,
                              hipStream_t stream) {
  (void)in_sizes; (void)n_in; (void)out_size; (void)ws_size;
  const float* x       = (const float*)d_in[0];
  const int*   lengths = (const int*)d_in[1];
  const float* Wi0     = (const float*)d_in[2];
  const float* Ws0     = (const float*)d_in[3];
  const float* bs0     = (const float*)d_in[4];
  const float* Wi1     = (const float*)d_in[5];
  const float* Ws1     = (const float*)d_in[6];
  const float* bs1     = (const float*)d_in[7];
  float* out = (float*)d_out;

  const size_t MT = (size_t)Bq * Tq;
  const size_t PI = MT * 6 * Hq;
  const size_t Y0 = MT * Hq;
  const size_t BH = (size_t)Bq * Hq;

  // ---- workspace carve ----
  char* w = (char*)d_ws;
  float*  pi   = (float*)w;   w += PI * sizeof(float);
  __bf16* xb   = (__bf16*)w;  w += MT * Dq * sizeof(__bf16);
  __bf16* y0b  = (__bf16*)w;  w += Y0 * sizeof(__bf16);
  __bf16* Wi0t = (__bf16*)w;  w += (size_t)6 * Hq * Dq * sizeof(__bf16);
  __bf16* Wi1t = (__bf16*)w;  w += (size_t)6 * Hq * Hq * sizeof(__bf16);
  __bf16* Ws0t = (__bf16*)w;  w += (size_t)5 * Hq * Hq * sizeof(__bf16);
  __bf16* Ws1t = (__bf16*)w;  w += (size_t)5 * Hq * Hq * sizeof(__bf16);
  // state block (zeroed together before each layer): hbf, hfp, c, counter
  char* state = w;
  __bf16* hbf = (__bf16*)w;   w += 2 * BH * sizeof(__bf16);
  float*  hfp = (float*)w;    w += 2 * BH * sizeof(float);
  float*  cb  = (float*)w;    w += BH * sizeof(float);
  int*    cnt = (int*)w;      w += 16;
  const size_t state_floats = ((size_t)(w - state)) / sizeof(float);

  float* y1 = out;            // (B,T,H)
  float* fh = out + Y0;       // (2,B,H)
  float* fc = fh + 2 * BH;    // (2,B,H)

  // allow >64KB dynamic LDS for the persistent kernel (idempotent, capture-safe)
  (void)hipFuncSetAttribute((const void*)lstm_persistent_kernel,
                            hipFuncAttributeMaxDynamicSharedMemorySize, (int)SMEM_PERS);

  // ---- one-time conversions ----
  {
    size_t n = MT * Dq;
    cvt_bf16_kernel<<<(int)((n / 4 + 255) / 256), 256, 0, stream>>>(x, xb, n);
    size_t nwi = (size_t)Dq * 6 * Hq;
    transpose_cvt_kernel<<<(int)((nwi + 255) / 256), 256, 0, stream>>>(Wi0, Wi0t, Dq, 6 * Hq);
    size_t nwi1 = (size_t)Hq * 6 * Hq;
    transpose_cvt_kernel<<<(int)((nwi1 + 255) / 256), 256, 0, stream>>>(Wi1, Wi1t, Hq, 6 * Hq);
    size_t nws = (size_t)Hq * 5 * Hq;
    transpose_cvt_kernel<<<(int)((nws + 255) / 256), 256, 0, stream>>>(Ws0, Ws0t, Hq, 5 * Hq);
    transpose_cvt_kernel<<<(int)((nws + 255) / 256), 256, 0, stream>>>(Ws1, Ws1t, Hq, 5 * Hq);
  }

  const dim3 gemm_grid(6 * Hq / 128, (int)(MT / 128));
  const int zstate_blocks = (int)((state_floats + 255) / 256);
  const int zcopy_blocks  = (int)((BH + 255) / 256);

  // ---- layer 0 ----
  lstm_gemm_kernel<<<gemm_grid, 256, 0, stream>>>(xb, Wi0t, pi, (int)MT, 6 * Hq, Dq);
  zero_kernel<<<zstate_blocks, 256, 0, stream>>>((float*)state, state_floats);
  lstm_persistent_kernel<<<NBLK, 256, SMEM_PERS, stream>>>(
      hbf, hfp, cb, pi, Ws0t, bs0, lengths, (float*)nullptr, y0b, cnt);
  copy_kernel<<<zcopy_blocks, 256, 0, stream>>>(hfp, fh, (int)BH);  // T even -> buf0
  copy_kernel<<<zcopy_blocks, 256, 0, stream>>>(cb,  fc, (int)BH);

  // ---- layer 1 ----
  lstm_gemm_kernel<<<gemm_grid, 256, 0, stream>>>(y0b, Wi1t, pi, (int)MT, 6 * Hq, Hq);
  zero_kernel<<<zstate_blocks, 256, 0, stream>>>((float*)state, state_floats);
  lstm_persistent_kernel<<<NBLK, 256, SMEM_PERS, stream>>>(
      hbf, hfp, cb, pi, Ws1t, bs1, lengths, y1, (__bf16*)nullptr, cnt);
  copy_kernel<<<zcopy_blocks, 256, 0, stream>>>(hfp, fh + BH, (int)BH);
  copy_kernel<<<zcopy_blocks, 256, 0, stream>>>(cb,  fc + BH, (int)BH);
}
